// Head_16939351015571
// MI455X (gfx1250) — matrix operand order repaired and checked by reading, not compiled, not run on hardware
//
#include <hip/hip_runtime.h>
#include <hip/hip_bf16.h>
#include <stdint.h>

// Problem dims (match reference)
#define EMBED 1024
#define HEAD  64
#define CTX   2048
#define BATCH 8

// WMMA fragment types (CDNA5 / gfx1250, wave32)
typedef __attribute__((ext_vector_type(16))) __bf16 v16bf;
typedef __attribute__((ext_vector_type(8)))  float  v8f;
typedef __attribute__((ext_vector_type(4)))  int    v4i;

union Frag16 {           // 8 VGPRs = 16 bf16 elements per lane
    uint32_t u[8];
    uint4    q[2];
    v16bf    v;
};

// ---- f32 -> bf16 pair pack: prefer v_cvt_pk_bf16_f32 when available ----
#if __has_builtin(__builtin_amdgcn_cvt_pk_bf16_f32)
__device__ __forceinline__ uint32_t pk_bf16(float lo, float hi) {
    auto r = __builtin_amdgcn_cvt_pk_bf16_f32(lo, hi);
    return __builtin_bit_cast(uint32_t, r);
}
#else
__device__ __forceinline__ uint32_t pk_bf16(float lo, float hi) {
    uint32_t a = __float_as_uint(lo), b = __float_as_uint(hi);
    a = (a + 0x7FFFu + ((a >> 16) & 1u)) >> 16;
    b = (b + 0x7FFFu + ((b >> 16) & 1u)) >> 16;
    return (a & 0xFFFFu) | (b << 16);
}
#endif
__device__ __forceinline__ uint16_t bf16h(float f) {
    uint32_t a = __float_as_uint(f);
    return (uint16_t)((a + 0x7FFFu + ((a >> 16) & 1u)) >> 16);
}

// ---- 16-byte global -> LDS stage: async tensor path when available ----
#if __has_builtin(__builtin_amdgcn_global_load_async_to_lds_b128)
#define HAVE_ASYNC_LDS 1
#endif

__device__ __forceinline__ void stage16(const void* g, void* l) {
#ifdef HAVE_ASYNC_LDS
    __builtin_amdgcn_global_load_async_to_lds_b128((v4i*)g, (v4i*)l, 0, 0);
#else
    *(uint4*)l = *(const uint4*)g;
#endif
}
__device__ __forceinline__ void stage_wait() {
#ifdef HAVE_ASYNC_LDS
#if __has_builtin(__builtin_amdgcn_s_wait_asynccnt)
    __builtin_amdgcn_s_wait_asynccnt(0);
#else
    asm volatile("s_wait_asynccnt 0x0" ::: "memory");
#endif
#endif
}

// ---------------------------------------------------------------------------
// Kernel 0: swizzle Wq/Wk/Wv (fp32 [1024][64]) into WMMA B-matrix layout:
//   wswz[mat][kb][nt][lane][j]  (uint32 = packed bf16 pair)
//   B 32x16 bf16, contiguous k order (same as A in qkv_kernel): lane(g,n) dword j holds W[kb*32 + g*16 + 2j + {0,1}][nt*16+n]
// ---------------------------------------------------------------------------
__global__ void wswz_kernel(const float* __restrict__ Wq,
                            const float* __restrict__ Wk,
                            const float* __restrict__ Wv,
                            uint32_t* __restrict__ wswz) {
    int idx  = blockIdx.x * 256 + threadIdx.x;        // 98304 total
    int j    = idx & 7;
    int lane = (idx >> 3) & 31;
    int nt   = (idx >> 8) & 3;
    int kb   = (idx >> 10) & 31;
    int mat  = idx >> 15;
    int g = lane >> 4, n = lane & 15;
    int k0  = kb * 32 + g * 16 + 2 * j;
    int col = nt * 16 + n;
    const float* W = (mat == 0) ? Wq : (mat == 1) ? Wk : Wv;
    wswz[idx] = pk_bf16(W[k0 * HEAD + col], W[(k0 + 1) * HEAD + col]);
}

// ---------------------------------------------------------------------------
// Kernel 1: QKV projection.  One wave -> 16 rows x 64 cols of one of {Q,K,V}.
// K-loop: 32 steps of v_wmma_f32_16x16x32_bf16, A = x tile (fp32->bf16 cvt),
// B = pre-swizzled weights.  Outputs:
//   qbf: bf16 [BT][64] row-major, pre-scaled by 1/sqrt(64)
//   kbf: bf16 [BT][64] row-major            (== QK^T B-fragment layout)
//   vtb: bf16 [B][64][2048] (V transposed)  (== PV   B-fragment layout)
// ---------------------------------------------------------------------------
__global__ void __launch_bounds__(128)
qkv_kernel(const float* __restrict__ x,
           const float* __restrict__ bq,
           const float* __restrict__ bk,
           const float* __restrict__ bv,
           const uint32_t* __restrict__ wswz,
           uint16_t* __restrict__ qbf,
           uint16_t* __restrict__ kbf,
           uint16_t* __restrict__ vtb) {
    int w    = blockIdx.x * 4 + (threadIdx.x >> 5);   // 3072 waves
    int lane = threadIdx.x & 31;
    int mat  = w >> 10;                               // 0=Q 1=K 2=V
    int tile = w & 1023;
    int rows0 = tile * 16;
    int g = lane >> 4, m = lane & 15, n = m;

    const uint32_t* wb   = wswz + mat * 32768;
    const float*    bias = (mat == 0) ? bq : (mat == 1) ? bk : bv;

    v8f c[4];
    #pragma unroll
    for (int nt = 0; nt < 4; ++nt) c[nt] = (v8f){0.f,0.f,0.f,0.f,0.f,0.f,0.f,0.f};

    const float* xr = x + (size_t)(rows0 + m) * EMBED;   // this lane's A row

    for (int kb = 0; kb < 32; ++kb) {
        // A fragment: contiguous k order matching wswz; lane(g,m) holds cols
        //   kb*32 + 16g + 0..15
        int base = kb * 32 + 16 * g;
        float4 a0 = *(const float4*)(xr + base);
        float4 a1 = *(const float4*)(xr + base + 4);
        float4 a2 = *(const float4*)(xr + base + 8);
        float4 a3 = *(const float4*)(xr + base + 12);
        Frag16 A;
        A.u[0] = pk_bf16(a0.x, a0.y);  A.u[1] = pk_bf16(a0.z, a0.w);
        A.u[2] = pk_bf16(a1.x, a1.y);  A.u[3] = pk_bf16(a1.z, a1.w);
        A.u[4] = pk_bf16(a2.x, a2.y);  A.u[5] = pk_bf16(a2.z, a2.w);
        A.u[6] = pk_bf16(a3.x, a3.y);  A.u[7] = pk_bf16(a3.z, a3.w);
        #pragma unroll
        for (int nt = 0; nt < 4; ++nt) {
            const uint4* bp = (const uint4*)(wb + ((size_t)(kb * 4 + nt) * 32 + lane) * 8);
            Frag16 B; B.q[0] = bp[0]; B.q[1] = bp[1];
            c[nt] = __builtin_amdgcn_wmma_f32_16x16x32_bf16(
                        false, A.v, false, B.v, (short)0, c[nt], false, false);
        }
    }

    // Epilogue: bias (+ q scale), fp32 -> bf16, scatter to layout-matched store.
    #pragma unroll
    for (int nt = 0; nt < 4; ++nt) {
        float bval = bias[nt * 16 + n];
        #pragma unroll
        for (int r = 0; r < 8; ++r) {
            float val = c[nt][r] + bval;
            if (mat == 0) val *= 0.125f;              // 1/sqrt(HEAD)
            int row = rows0 + r + 8 * g;              // C layout: M = r + 8g
            int col = nt * 16 + n;
            if (mat == 0)      qbf[(size_t)row * HEAD + col] = bf16h(val);
            else if (mat == 1) kbf[(size_t)row * HEAD + col] = bf16h(val);
            else {
                int bb = row >> 11, t = row & 2047;
                vtb[((size_t)(bb * HEAD + col)) * CTX + t] = bf16h(val);
            }
        }
    }
}

// ---------------------------------------------------------------------------
// Kernel 2: flash attention.  Block = 4 waves = 4 consecutive query tiles of
// the same batch.  All waves run a uniform step count (extra steps are fully
// masked -> exact no-ops), so the block can cooperatively stage each 32-key
// K/V tile into LDS (async global->LDS when available) and share it 4 ways.
// ---------------------------------------------------------------------------
__global__ void __launch_bounds__(128)
attn_kernel(const uint16_t* __restrict__ qbf,
            const uint16_t* __restrict__ kbf,
            const uint16_t* __restrict__ vtb,
            float* __restrict__ out) {
    __shared__ __align__(16) uint16_t klds[32 * 64];        // [key][h]   4KB
    __shared__ __align__(16) uint16_t vlds[64 * 32];        // [h][key]   4KB
    __shared__ __align__(16) uint16_t plds_all[4][16 * 32]; // per-wave P 4KB
    int tid  = threadIdx.x;
    int widx = tid >> 5;
    int lane = tid & 31;
    uint16_t* plds = plds_all[widx];

    int g = lane >> 4, m = lane & 15, n = m;
    int wq = blockIdx.x * 4 + widx;                   // qtile id 0..1023
    int b  = wq >> 7;                                 // same for whole block
    int qt = wq & 127;
    int qt_hi = (blockIdx.x * 4 + 3) & 127;           // block's max qtile
    int nsteps = ((qt_hi * 16 + 15) >> 5) + 1;        // uniform causal bound

    const uint16_t* qb = qbf + (size_t)b * CTX * HEAD;
    const uint16_t* kB = kbf + (size_t)b * CTX * HEAD;
    const uint16_t* vt = vtb + (size_t)b * HEAD * CTX;

    // Q A-fragments, h-blocks 0..31 and 32..63 (loaded once)
    Frag16 aq0, aq1;
    {
        const uint16_t* qr = qb + (size_t)(qt * 16 + m) * HEAD;
        aq0.q[0] = *(const uint4*)(qr +       8 * g);
        aq0.q[1] = *(const uint4*)(qr + 16 +  8 * g);
        aq1.q[0] = *(const uint4*)(qr + 32 +  8 * g);
        aq1.q[1] = *(const uint4*)(qr + 48 +  8 * g);
    }

    float mi[8], li[8];
    #pragma unroll
    for (int r = 0; r < 8; ++r) { mi[r] = -__builtin_inff(); li[r] = 0.f; }
    v8f o[4];
    #pragma unroll
    for (int nt = 0; nt < 4; ++nt) o[nt] = (v8f){0.f,0.f,0.f,0.f,0.f,0.f,0.f,0.f};

    for (int kt = 0; kt < nsteps; ++kt) {
        int kb32 = kt * 32;

        // ---- cooperative stage of K (32x64) and V^T (64x32) tiles ----
        // K source is 4KB contiguous; V source is 64 rows x 64B (stride 4KB).
        {
            const char* ksrc = (const char*)(kB + (size_t)kb32 * HEAD);
            #pragma unroll
            for (int c = 0; c < 2; ++c) {
                int ck = tid + c * 128;               // 0..255, 16B chunks
                stage16(ksrc + ck * 16, (char*)klds + ck * 16);
                int h = ck >> 2, part = ck & 3;
                stage16((const char*)(vt + (size_t)h * CTX + kb32 + part * 8),
                        (char*)vlds + h * 64 + part * 16);
            }
            stage_wait();
            __syncthreads();
        }

        // ---- S = Q K^T (pre-scaled), causal mask ----
        v8f s[2];
        #pragma unroll
        for (int sub = 0; sub < 2; ++sub) {
            const uint16_t* kr = klds + (sub * 16 + n) * 64;   // key row, 128B
            Frag16 bk0, bk1;                                   // h 0..31 / 32..63
            bk0.q[0] = *(const uint4*)(kr +       8 * g);
            bk0.q[1] = *(const uint4*)(kr + 16 +  8 * g);
            bk1.q[0] = *(const uint4*)(kr + 32 +  8 * g);
            bk1.q[1] = *(const uint4*)(kr + 48 +  8 * g);
            v8f t = (v8f){0.f,0.f,0.f,0.f,0.f,0.f,0.f,0.f};
            t = __builtin_amdgcn_wmma_f32_16x16x32_bf16(
                    false, aq0.v, false, bk0.v, (short)0, t, false, false);
            t = __builtin_amdgcn_wmma_f32_16x16x32_bf16(
                    false, aq1.v, false, bk1.v, (short)0, t, false, false);
            int key = kb32 + sub * 16 + n;
            #pragma unroll
            for (int r = 0; r < 8; ++r) {
                int qrow = qt * 16 + r + 8 * g;       // C layout: M = r + 8g
                t[r] = (key <= qrow) ? t[r] : -__builtin_inff();
            }
            s[sub] = t;
        }

        // ---- online softmax: row max / rescale / exp / row sum ----
        // (fully-masked steps degrade to mn=mi, sc=1, p=0 -> exact no-op)
        float mn[8], sc[8];
        #pragma unroll
        for (int r = 0; r < 8; ++r) {
            float t = fmaxf(s[0][r], s[1][r]);
            t = fmaxf(t, __shfl_xor(t, 1, 32));
            t = fmaxf(t, __shfl_xor(t, 2, 32));
            t = fmaxf(t, __shfl_xor(t, 4, 32));
            t = fmaxf(t, __shfl_xor(t, 8, 32));       // rows live in 16-lane halves
            mn[r] = fmaxf(mi[r], t);
            sc[r] = __expf(mi[r] - mn[r]);
            mi[r] = mn[r];
        }
        #pragma unroll
        for (int r = 0; r < 8; ++r) {
            float p0 = __expf(s[0][r] - mn[r]);
            float p1 = __expf(s[1][r] - mn[r]);
            plds[(r + 8 * g) * 32 +      n] = bf16h(p0);   // P tile, row-major
            plds[(r + 8 * g) * 32 + 16 + n] = bf16h(p1);
            float rs = p0 + p1;
            rs += __shfl_xor(rs, 1, 32);
            rs += __shfl_xor(rs, 2, 32);
            rs += __shfl_xor(rs, 4, 32);
            rs += __shfl_xor(rs, 8, 32);
            li[r] = li[r] * sc[r] + rs;
            #pragma unroll
            for (int nt = 0; nt < 4; ++nt) o[nt][r] *= sc[r];
        }

        // intra-wave LDS store->load ordering (DS is in-order per wave)
        asm volatile("s_wait_dscnt 0x0" ::: "memory");

        // ---- P A-fragment from LDS, then O += P V ----
        Frag16 pA;
        pA.q[0] = *(const uint4*)(plds + m * 32 +      8 * g);
        pA.q[1] = *(const uint4*)(plds + m * 32 + 16 + 8 * g);
        #pragma unroll
        for (int nt = 0; nt < 4; ++nt) {
            const uint16_t* vr = vlds + (nt * 16 + n) * 32 + 8 * g;  // h row, 64B
            Frag16 bv; bv.q[0] = *(const uint4*)(vr); bv.q[1] = *(const uint4*)(vr + 16);
            o[nt] = __builtin_amdgcn_wmma_f32_16x16x32_bf16(
                        false, pA.v, false, bv.v, (short)0, o[nt], false, false);
        }

        if (kt + 1 < nsteps)   // pull next K tile toward L2
            __builtin_prefetch((const void*)(kB + (size_t)(kb32 + 32) * HEAD + tid * 16), 0, 1);

        __syncthreads();       // all waves done reading K/V LDS before restage
    }

    // ---- epilogue: O / l, fp32 store ----
    #pragma unroll
    for (int r = 0; r < 8; ++r) {
        float inv = 1.0f / li[r];
        int qrow = qt * 16 + r + 8 * g;
        #pragma unroll
        for (int nt = 0; nt < 4; ++nt)
            out[(size_t)(b * CTX + qrow) * HEAD + nt * 16 + n] = o[nt][r] * inv;
    }
}

// ---------------------------------------------------------------------------
// Launch: setup -> projection -> attention, all on `stream`.
// Workspace layout (bytes):
//   [0, 393216)        wswz  (3*32*4*32*8 dwords)
//   [393216, +2MB)     qbf   bf16 [16384][64]
//   [2490368, +2MB)    kbf   bf16 [16384][64]
//   [4587520, +2MB)    vtb   bf16 [8][64][2048]
// ---------------------------------------------------------------------------
extern "C" void kernel_launch(void* const* d_in, const int* in_sizes, int n_in,
                              void* d_out, int out_size, void* d_ws, size_t ws_size,
                              hipStream_t stream) {
    const float* x  = (const float*)d_in[0];
    const float* Wq = (const float*)d_in[1];
    const float* bq = (const float*)d_in[2];
    const float* Wk = (const float*)d_in[3];
    const float* bk = (const float*)d_in[4];
    const float* Wv = (const float*)d_in[5];
    const float* bv = (const float*)d_in[6];

    char* ws = (char*)d_ws;
    uint32_t* wswz = (uint32_t*)(ws);
    uint16_t* qbf  = (uint16_t*)(ws + 393216);
    uint16_t* kbf  = (uint16_t*)(ws + 2490368);
    uint16_t* vtb  = (uint16_t*)(ws + 4587520);
    float* out = (float*)d_out;

    wswz_kernel<<<384, 256, 0, stream>>>(Wq, Wk, Wv, wswz);
    qkv_kernel<<<768, 128, 0, stream>>>(x, bq, bk, bv, wswz, qbf, kbf, vtb);
    attn_kernel<<<256, 128, 0, stream>>>(qbf, kbf, vtb, out);
}
